// TE_43482248905360
// MI455X (gfx1250) — matrix operand order repaired and checked by reading, not compiled, not run on hardware
//
#include <hip/hip_runtime.h>

// ---------------------------------------------------------------------------
// Types for CDNA5 WMMA bf16 path: v_wmma_f32_16x16x32_bf16
// ---------------------------------------------------------------------------
typedef __bf16 bf16_t;
typedef __attribute__((ext_vector_type(16))) __bf16 v16bf;
typedef __attribute__((ext_vector_type(8)))  __bf16 v8bf;
typedef __attribute__((ext_vector_type(8)))  float  v8f;

#define D_    256
#define GATES 1024
#define BM    32          // batch rows per block (2 WMMA row-tiles)

union ABfrag { v16bf v; v8bf h[2]; };

// Describes where timestep t of an LSTM input sequence comes from.
// mode 0: broadcast vector  (base[0:256])
// mode 1: gather            base[idx[row*stride+off]*256 + :]
// mode 2: direct strided    base[(row*stride+off)*256 + :]
struct StepDesc {
    const float* base;
    const int*   idx;
    int          stride;
    int          off;
    int          mode;
};
struct SeqDesc { StepDesc s[8]; };

__device__ __forceinline__ float sigm(float x) { return 1.f / (1.f + __expf(-x)); }

// ---------------------------------------------------------------------------
// Fused gather + LSTM.  Block = 32 batch rows, 8 waves.  Wave w computes
// z[0:32, w*128 : (w+1)*128] as 2 row-tiles x 8 col-tiles of 16x16x32 bf16
// WMMA (K=256 contraction); every weight tile is reused for both row-tiles,
// halving L2 weight traffic vs a 16-row block (arith intensity ~67 flop/B).
// Cell state c and hidden h (f32) live in per-thread registers across steps.
// Weights pre-swizzled: tile(kc,ct) = 32x16, lane 16*((k/8)&1)+n%16 holds 16 K values.
// ---------------------------------------------------------------------------
template<int T>
__global__ __launch_bounds__(256)
void lstm_wmma_kernel(SeqDesc seq,
                      const bf16_t* __restrict__ Wk,   // [256,1024] swizzled
                      const bf16_t* __restrict__ Wr,   // [256,1024] swizzled
                      const float*  __restrict__ bias, // [1024]
                      float* __restrict__ out, int N)
{
    __shared__ bf16_t xs[BM][D_];     // x_t, bf16           (16 KB)
    __shared__ bf16_t hs[BM][D_];     // h_{t-1}, bf16       (16 KB)
    __shared__ float  zs[BM][GATES];  // gate pre-act        (128 KB)

    const int tid  = threadIdx.x;
    const int lane = tid & 31;
    const int wave = tid >> 5;
    const int r0   = blockIdx.x * BM;

    // stage/gate mapping: thread owns row gm, cols [gn0, gn0+32)
    const int gm  = tid >> 3;
    const int gn0 = (tid & 7) * 32;

    float creg[32];                   // persistent cell state
    float hreg[32];                   // persistent hidden state (f32)
    #pragma unroll
    for (int e = 0; e < 32; ++e) { creg[e] = 0.f; hreg[e] = 0.f; }
    #pragma unroll
    for (int e = 0; e < 32; ++e) hs[gm][gn0 + e] = (bf16_t)0.f;

    const int M      = lane & 15;
    const int Ks     = (lane >= 16) ? 8 : 0;   // 16-bit A-matrix lane layout
    const int rowsel = (lane >= 16) ? 8 : 0;   // 32-bit C/D lane layout
    const int nsel   = lane & 15;
    const int colTile0 = wave * 8;

    for (int t = 0; t < T; ++t) {
        { // stage x_t into LDS (gather + f32->bf16)
            int gr = r0 + gm; int grc = (gr < N) ? gr : 0;
            const StepDesc sd = seq.s[t];
            const float* src;
            if (sd.mode == 0)      src = sd.base;
            else if (sd.mode == 1) { int id = sd.idx[grc * sd.stride + sd.off];
                                     src = sd.base + (size_t)id * D_; }
            else                   src = sd.base + ((size_t)grc * sd.stride + sd.off) * (size_t)D_;
            #pragma unroll
            for (int e = 0; e < 32; ++e) xs[gm][gn0 + e] = (bf16_t)src[gn0 + e];
        }
        __syncthreads();

        v8f acc[2][8] = {};
        #pragma unroll
        for (int kc = 0; kc < 8; ++kc) {
            ABfrag ax[2], ah[2];
            #pragma unroll
            for (int rt = 0; rt < 2; ++rt) {
                ax[rt].h[0] = *(const v8bf*)&xs[rt * 16 + M][kc * 32 + Ks];
                ax[rt].h[1] = *(const v8bf*)&xs[rt * 16 + M][kc * 32 + 16 + Ks];
                ah[rt].h[0] = *(const v8bf*)&hs[rt * 16 + M][kc * 32 + Ks];
                ah[rt].h[1] = *(const v8bf*)&hs[rt * 16 + M][kc * 32 + 16 + Ks];
            }
            if (kc < 7) { // prefetch next K-chunk of the weight stream
                size_t pf = ((size_t)((kc + 1) * 64 + colTile0) * 512) + (size_t)lane * 16;
                __builtin_prefetch(Wk + pf, 0, 0);
                __builtin_prefetch(Wr + pf, 0, 0);
            }
            #pragma unroll
            for (int ct = 0; ct < 8; ++ct) {
                size_t toff = ((size_t)(kc * 64 + colTile0 + ct) * 512) + (size_t)lane * 16;
                v16bf bk = *(const v16bf*)(Wk + toff);
                v16bf br = *(const v16bf*)(Wr + toff);
                #pragma unroll
                for (int rt = 0; rt < 2; ++rt) {
                    acc[rt][ct] = __builtin_amdgcn_wmma_f32_16x16x32_bf16(
                                      false, ax[rt].v, false, bk, (short)0,
                                      acc[rt][ct], false, false);
                    acc[rt][ct] = __builtin_amdgcn_wmma_f32_16x16x32_bf16(
                                      false, ah[rt].v, false, br, (short)0,
                                      acc[rt][ct], false, false);
                }
            }
        }
        #pragma unroll
        for (int rt = 0; rt < 2; ++rt)          // spill z slab to LDS
            #pragma unroll
            for (int ct = 0; ct < 8; ++ct) {
                int col = wave * 128 + ct * 16 + nsel;
                #pragma unroll
                for (int v = 0; v < 8; ++v)
                    zs[rt * 16 + v + rowsel][col] = acc[rt][ct][v];
            }
        __syncthreads();

        { // elementwise gates (Keras order i,f,g,o); c,h stay in registers
            #pragma unroll
            for (int e = 0; e < 32; ++e) {
                int n = gn0 + e;
                float ig = sigm (zs[gm][n]       + bias[n]);
                float fg = sigm (zs[gm][256 + n] + bias[256 + n]);
                float gg = tanhf(zs[gm][512 + n] + bias[512 + n]);
                float og = sigm (zs[gm][768 + n] + bias[768 + n]);
                float c  = fg * creg[e] + ig * gg;
                creg[e]  = c;
                float h  = og * tanhf(c);
                hreg[e]  = h;
                hs[gm][n] = (bf16_t)h;
            }
        }
        __syncthreads();
    }

    { // final hidden state -> global (f32 from registers)
        int gr = r0 + gm;
        if (gr < N) {
            #pragma unroll
            for (int e = 0; e < 32; ++e) out[(size_t)gr * D_ + gn0 + e] = hreg[e];
        }
    }
}

// ---------------------------------------------------------------------------
// fe = concat(h, table[ret]) @ w_fn   ([N,512] x [512,256] -> [N,256])
// Safe to run in place over hp: a block only touches its own 16 rows.
// ---------------------------------------------------------------------------
__global__ __launch_bounds__(256)
void fngemm_kernel(const float* __restrict__ hp,
                   const float* __restrict__ table,
                   const int*   __restrict__ ret,
                   const bf16_t* __restrict__ Wt,    // [512,256] swizzled
                   float* __restrict__ out, int N)
{
    __shared__ bf16_t as_[16][512];
    const int tid = threadIdx.x, lane = tid & 31, wave = tid >> 5;
    const int r0 = blockIdx.x * 16;
    {
        int m = tid >> 4, d0 = (tid & 15) * 32;
        int gr = r0 + m, grc = (gr < N) ? gr : 0;
        const float* h  = hp    + (size_t)grc * 256;
        const float* rr = table + (size_t)ret[grc] * 256;
        #pragma unroll
        for (int e = 0; e < 32; ++e) {
            int d = d0 + e;
            as_[m][d] = (bf16_t)((d < 256) ? h[d] : rr[d - 256]);
        }
    }
    __syncthreads();

    const int M = lane & 15, Ks = (lane >= 16) ? 8 : 0;
    v8f acc[2] = {};
    #pragma unroll
    for (int kc = 0; kc < 16; ++kc) {
        ABfrag a;
        a.h[0] = *(const v8bf*)&as_[M][kc * 32 + Ks];
        a.h[1] = *(const v8bf*)&as_[M][kc * 32 + 16 + Ks];
        #pragma unroll
        for (int ct = 0; ct < 2; ++ct) {
            size_t toff = ((size_t)(kc * 16 + wave * 2 + ct) * 512) + (size_t)lane * 16;
            v16bf b = *(const v16bf*)(Wt + toff);
            acc[ct] = __builtin_amdgcn_wmma_f32_16x16x32_bf16(
                          false, a.v, false, b, (short)0, acc[ct], false, false);
        }
    }
    int rowsel = (lane >= 16) ? 8 : 0, nsel = lane & 15;
    #pragma unroll
    for (int ct = 0; ct < 2; ++ct) {
        int col = wave * 32 + ct * 16 + nsel;
        #pragma unroll
        for (int v = 0; v < 8; ++v) {
            int gr = r0 + rowsel + v;
            if (gr < N) out[(size_t)gr * 256 + col] = acc[ct][v];
        }
    }
}

// ---------------------------------------------------------------------------
// f32 [K,N] weight -> bf16 WMMA-B swizzled tiles: tile = (k/32)*(N/16)+(n/16),
// within tile: lane = 16*((k%32/8)&1) + n%16 holds 16 K values in table order (one 32B load/lane).
// ---------------------------------------------------------------------------
__global__ void convert_swizzle_kernel(const float* __restrict__ W,
                                       bf16_t* __restrict__ out, int K, int Ncols)
{
    int i = blockIdx.x * blockDim.x + threadIdx.x;
    if (i >= K * Ncols) return;
    int k = i / Ncols, n = i % Ncols;
    int tile = (k >> 5) * (Ncols >> 4) + (n >> 4);
    out[(size_t)tile * 512 + (size_t)((((k >> 3) & 1) * 16) + (n & 15)) * 16 + (k & 7) + (((k & 31) >> 4) << 3)] = (bf16_t)W[i];
}

__global__ void gather_basic_kernel(const float* __restrict__ bt,
                                    const int* __restrict__ bidx,
                                    float* __restrict__ table, int N)
{
    int i = blockIdx.x * blockDim.x + threadIdx.x;
    if (i >= N * 256) return;
    int r = i >> 8, d = i & 255;
    table[i] = bt[(size_t)bidx[r] * 256 + d];
}

// ---------------------------------------------------------------------------
extern "C" void kernel_launch(void* const* d_in, const int* in_sizes, int n_in,
                              void* d_out, int out_size, void* d_ws, size_t ws_size,
                              hipStream_t stream)
{
    (void)in_sizes; (void)n_in; (void)out_size; (void)ws_size;

    const float* kp = (const float*)d_in[1];
    const float* rp = (const float*)d_in[2];
    const float* bp = (const float*)d_in[3];
    const float* km = (const float*)d_in[4];
    const float* rm = (const float*)d_in[5];
    const float* bm = (const float*)d_in[6];
    const float* ks = (const float*)d_in[7];
    const float* rs = (const float*)d_in[8];
    const float* bs = (const float*)d_in[9];
    const float* empty_params  = (const float*)d_in[10];
    const float* empty_members = (const float*)d_in[11];
    const float* w_fn = (const float*)d_in[12];

    const int N0 = 50000, NL = 50000, Mtop = 50000;
    const int P = 4, F = 2, Kp = 3;

    // workspace carve-out (256B aligned regions)
    char* w = (char*)d_ws;
    auto alloc = [&](size_t bytes) -> char* {
        char* p = w; w += (bytes + 255) & ~(size_t)255; return p;
    };
    bf16_t* kp_t  = (bf16_t*)alloc((size_t)256 * 1024 * 2);
    bf16_t* rp_t  = (bf16_t*)alloc((size_t)256 * 1024 * 2);
    bf16_t* km_t  = (bf16_t*)alloc((size_t)256 * 1024 * 2);
    bf16_t* rm_t  = (bf16_t*)alloc((size_t)256 * 1024 * 2);
    bf16_t* ks_t  = (bf16_t*)alloc((size_t)256 * 1024 * 2);
    bf16_t* rs_t  = (bf16_t*)alloc((size_t)256 * 1024 * 2);
    bf16_t* wfn_t = (bf16_t*)alloc((size_t)512 * 256 * 2);
    float*  hp    = (float*)alloc((size_t)NL * F * 256 * 4);  // h / fe (in place)
    float*  hmem  = (float*)alloc((size_t)NL * 256 * 4);

    convert_swizzle_kernel<<<(256 * 1024 + 255) / 256, 256, 0, stream>>>(kp, kp_t, 256, 1024);
    convert_swizzle_kernel<<<(256 * 1024 + 255) / 256, 256, 0, stream>>>(rp, rp_t, 256, 1024);
    convert_swizzle_kernel<<<(256 * 1024 + 255) / 256, 256, 0, stream>>>(km, km_t, 256, 1024);
    convert_swizzle_kernel<<<(256 * 1024 + 255) / 256, 256, 0, stream>>>(rm, rm_t, 256, 1024);
    convert_swizzle_kernel<<<(256 * 1024 + 255) / 256, 256, 0, stream>>>(ks, ks_t, 256, 1024);
    convert_swizzle_kernel<<<(256 * 1024 + 255) / 256, 256, 0, stream>>>(rs, rs_t, 256, 1024);
    convert_swizzle_kernel<<<(512 * 256  + 255) / 256, 256, 0, stream>>>(w_fn, wfn_t, 512, 256);

    float* table  = (float*)d_out;                       // [200000,256]
    float* fn_out = table + (size_t)(N0 + 3 * NL) * 256; // [50000,256]

    gather_basic_kernel<<<(N0 * 256 + 255) / 256, 256, 0, stream>>>(
        (const float*)d_in[0], (const int*)d_in[13], table, N0);

    for (int l = 0; l < 3; ++l) {
        const int* prop   = (const int*)d_in[14 + l * 4 + 0];
        const int* fparam = (const int*)d_in[14 + l * 4 + 1];
        const int* fret   = (const int*)d_in[14 + l * 4 + 2];
        const int* sup    = (const int*)d_in[14 + l * 4 + 3];
        const int NF = NL * F;

        SeqDesc sq{};                                     // func param LSTM, T=4
        sq.s[0] = { empty_params, nullptr, 0, 0, 0 };
        for (int t = 1; t <= 3; ++t) sq.s[t] = { table, fparam, Kp, t - 1, 1 };
        lstm_wmma_kernel<4><<<(NF + BM - 1) / BM, 256, 0, stream>>>(sq, kp_t, rp_t, bp, hp, NF);

        fngemm_kernel<<<(NF + 15) / 16, 256, 0, stream>>>(hp, table, fret, wfn_t, hp, NF);

        SeqDesc sm{};                                     // member LSTM, T=7
        sm.s[0] = { empty_members, nullptr, 0, 0, 0 };
        for (int t = 1; t <= 4; ++t) sm.s[t] = { table, prop, P, t - 1, 1 };
        sm.s[5] = { hp, nullptr, F, 0, 2 };
        sm.s[6] = { hp, nullptr, F, 1, 2 };
        lstm_wmma_kernel<7><<<(NL + BM - 1) / BM, 256, 0, stream>>>(sm, km_t, rm_t, bm, hmem, NL);

        SeqDesc ss{};                                     // super LSTM, T=3 -> table append
        ss.s[0] = { table, sup, 2, 0, 1 };
        ss.s[1] = { table, sup, 2, 1, 1 };
        ss.s[2] = { hmem, nullptr, 1, 0, 2 };
        lstm_wmma_kernel<3><<<(NL + BM - 1) / BM, 256, 0, stream>>>(
            ss, ks_t, rs_t, bs, table + (size_t)(N0 + l * NL) * 256, NL);
    }

    SeqDesc st{};                                         // top-level functions
    st.s[0] = { empty_params, nullptr, 0, 0, 0 };
    for (int t = 1; t <= 3; ++t) st.s[t] = { table, (const int*)d_in[26], Kp, t - 1, 1 };
    lstm_wmma_kernel<4><<<(Mtop + BM - 1) / BM, 256, 0, stream>>>(st, kp_t, rp_t, bp, hp, Mtop);
    fngemm_kernel<<<(Mtop + 15) / 16, 256, 0, stream>>>(hp, table, (const int*)d_in[27],
                                                        wfn_t, fn_out, Mtop);
}
